// GINModel_58437325029805
// MI455X (gfx1250) — matrix operand order repaired and checked by reading, not compiled, not run on hardware
//
#include <hip/hip_runtime.h>

#define N_NODES   100000
#define N_EDGES   1600000
#define NUM_GRAPHS 64
#define GA 6

typedef __attribute__((ext_vector_type(16))) __bf16 v16bf;
typedef __attribute__((ext_vector_type(8)))  float  v8f;

// Native 2xf32 -> packed bf16 (round-to-nearest-even in HW).
__device__ __forceinline__ unsigned int cvt_pk_bf16(float a, float b) {
    unsigned int r;
    asm("v_cvt_pk_bf16_f32 %0, %1, %2" : "=v"(r) : "v"(a), "v"(b));
    return r;
}

// Device-scope non-returning f32 atomic add (guaranteed native instruction).
__device__ __forceinline__ void gatomic_add_f32(float* addr, float v) {
    asm volatile("global_atomic_add_f32 %0, %1, off scope:SCOPE_DEV"
                 :: "v"(addr), "v"(v) : "memory");
}

// ---------------------------------------------------------------------------
// C[M x 256] = relu?( A[M x K] @ W[K x 256](bf16) + bias )
//   ABF16  : A is packed bf16 (lda=K elements) -> staged with
//            GLOBAL_LOAD_ASYNC_TO_LDS_B128 (ASYNCcnt), no conversion.
//   !ABF16 : A is f32 -> b128 loads + v_cvt_pk_bf16_f32 into LDS.
//   OUTBF16: epilogue stores bf16 (feeds the next GEMM), else f32.
// 256 threads = 8 wave32; block tile 128x128, K step 32; WMMA bf16 16x16x32.
// ---------------------------------------------------------------------------
template<bool ABF16, bool OUTBF16>
__global__ __launch_bounds__(256)
void gemm_wmma(const void* __restrict__ Ain, int lda, int K,
               const unsigned short* __restrict__ W,
               const float* __restrict__ bias,
               void* __restrict__ Cout, int M, int do_relu)
{
    __shared__ unsigned short As[128][40];   // [m][k], 80B rows: 16B-aligned, conflict-free
    __shared__ unsigned short Bs[128][40];   // [n][k] (transposed on load)

    const int tid  = threadIdx.x;
    const int lane = tid & 31;
    const int wid  = tid >> 5;
    const int half = lane >> 4;
    const int l16  = lane & 15;
    const int mwave = (wid & 3) * 32;   // 4 waves along M
    const int nwave = (wid >> 2) * 64;  // 2 waves along N
    const int m0 = blockIdx.x * 128;
    const int n0 = blockIdx.y * 128;

    v8f acc[2][4];
    for (int mt = 0; mt < 2; ++mt)
        for (int nt = 0; nt < 4; ++nt)
            for (int r = 0; r < 8; ++r) acc[mt][nt][r] = 0.0f;

    for (int k0 = 0; k0 < K; k0 += 32) {
        if (ABF16) {
            // ---- async DMA: 128x32 bf16 tile straight into LDS ----
            const unsigned short* Ab = (const unsigned short*)Ain;
            for (int t = 0; t < 2; ++t) {
                int i   = tid + t * 256;        // 16B chunk 0..511
                int row = i >> 2;               // 4 chunks per row
                int c8  = (i & 3) * 8;          // bf16 element offset
                int rg  = m0 + row;
                if (rg < M) {
                    unsigned int ldsa =
                        (unsigned int)(unsigned long long)(const void*)&As[row][c8];
                    const unsigned short* gp = Ab + (long long)rg * lda + k0 + c8;
                    asm volatile("global_load_async_to_lds_b128 %0, %1, off"
                                 :: "v"(ldsa), "v"(gp) : "memory");
                }
            }
        } else {
            // ---- 128x32 f32 tile -> bf16 via v_cvt_pk_bf16_f32 ----
            const float* Af = (const float*)Ain;
            for (int t = 0; t < 4; ++t) {
                int i   = tid + t * 256;        // float4 slot 0..1023
                int row = i >> 3;
                int c4  = (i & 7) * 4;
                float4 v = make_float4(0.f, 0.f, 0.f, 0.f);
                int rg = m0 + row;
                if (rg < M) {
                    const float* p = Af + (long long)rg * lda + k0 + c4;
                    v = *(const float4*)p;
                    if (k0 + 32 < K) __builtin_prefetch(p + 32, 0, 1);
                }
                *(unsigned int*)&As[row][c4]     = cvt_pk_bf16(v.x, v.y);
                *(unsigned int*)&As[row][c4 + 2] = cvt_pk_bf16(v.z, v.w);
            }
        }
        // ---- stage B tile (32x128 bf16, transposed to [n][k]) ----
        for (int t = 0; t < 8; ++t) {
            int i  = tid + t * 256;             // u32 slot 0..2047
            int k  = i >> 6;
            int n2 = (i & 63) * 2;
            unsigned int w = *(const unsigned int*)(W + (long long)(k0 + k) * 256 + n0 + n2);
            Bs[n2][k]     = (unsigned short)(w & 0xFFFFu);
            Bs[n2 + 1][k] = (unsigned short)(w >> 16);
        }
        if (ABF16) asm volatile("s_wait_asynccnt 0x0" ::: "memory");
        __syncthreads();

        // ---- assemble fragments per ISA 7.12.2 layouts ----
        union Frag { v16bf v; unsigned int u[8]; } af[2], bfr[4];
        for (int mt = 0; mt < 2; ++mt) {
            int row = mwave + mt * 16 + l16;
            for (int j = 0; j < 8; ++j) {
                int kk = ((j < 4) ? 0 : 16) + half * 8 + (j & 3) * 2;
                af[mt].u[j] = *(const unsigned int*)&As[row][kk];
            }
        }
        for (int nt = 0; nt < 4; ++nt) {
            int col = nwave + nt * 16 + l16;
            for (int j = 0; j < 8; ++j)
                bfr[nt].u[j] = *(const unsigned int*)&Bs[col][((j < 4) ? 0 : 16) + half * 8 + (j & 3) * 2];
        }
        for (int mt = 0; mt < 2; ++mt)
            for (int nt = 0; nt < 4; ++nt)
                acc[mt][nt] = __builtin_amdgcn_wmma_f32_16x16x32_bf16(
                    false, af[mt].v, false, bfr[nt].v,
                    (short)0, acc[mt][nt], false, false);
        __syncthreads();
    }

    // ---- epilogue: bias (+ReLU), store f32 or bf16 ----
    for (int mt = 0; mt < 2; ++mt) {
        for (int nt = 0; nt < 4; ++nt) {
            int colH = n0 + nwave + nt * 16 + l16;
            float b = bias[colH];
            for (int r = 0; r < 8; ++r) {
                int row = m0 + mwave + mt * 16 + r + 8 * half;
                if (row < M) {
                    float val = acc[mt][nt][r] + b;
                    if (do_relu) val = fmaxf(val, 0.0f);
                    if (OUTBF16) {
                        ((unsigned short*)Cout)[(long long)row * 256 + colH] =
                            (unsigned short)(cvt_pk_bf16(val, 0.0f) & 0xFFFFu);
                    } else {
                        ((float*)Cout)[(long long)row * 256 + colH] = val;
                    }
                }
            }
        }
    }
}

// out[dst[e]] += feat[src[e]] ; one thread per (edge, float4)
__global__ __launch_bounds__(256)
void scatter_add_kernel(const float* __restrict__ feat, const int* __restrict__ src,
                        const int* __restrict__ dst, float* __restrict__ out,
                        int shift, int mask, int dim)
{
    long long i = (long long)blockIdx.x * 256 + threadIdx.x;
    int e  = (int)(i >> shift);
    int c4 = ((int)i & mask) * 4;
    int s = src[e];
    int d = dst[e];
    float4 v = *(const float4*)(feat + (long long)s * dim + c4);
    float* o = out + (long long)d * dim + c4;
    gatomic_add_f32(o,     v.x);
    gatomic_add_f32(o + 1, v.y);
    gatomic_add_f32(o + 2, v.z);
    gatomic_add_f32(o + 3, v.w);
}

__global__ void copy_f4(const float4* __restrict__ in, float4* __restrict__ out, long long n4)
{
    long long i = (long long)blockIdx.x * 256 + threadIdx.x;
    if (i < n4) out[i] = in[i];
}

// pairs of f32 -> packed bf16 (weights; counts are even)
__global__ void f32_to_bf16_kernel(const float* __restrict__ in,
                                   unsigned int* __restrict__ out, int npairs)
{
    int i = blockIdx.x * 256 + threadIdx.x;
    if (i < npairs) out[i] = cvt_pk_bf16(in[2 * i], in[2 * i + 1]);
}

__global__ void zero_f32(float* p, int n)
{
    int i = blockIdx.x * 256 + threadIdx.x;
    if (i < n) p[i] = 0.0f;
}

// Segment-sum over sorted batch: 64-node window per block, register
// accumulation per column, atomics only at graph boundaries.
__global__ __launch_bounds__(256)
void pool_kernel(const float* __restrict__ h, const int* __restrict__ batch,
                 float* __restrict__ sums, float* __restrict__ cnts, int N)
{
    int n0 = blockIdx.x * 64;
    int nend = min(n0 + 64, N);
    int c = threadIdx.x;
    float acc = 0.f, cacc = 0.f;
    int cur = batch[n0];
    for (int n = n0; n < nend; ++n) {
        int b = batch[n];
        if (b != cur) {
            gatomic_add_f32(sums + (long long)cur * 256 + c, acc);
            if (c == 0) gatomic_add_f32(cnts + cur, cacc);
            acc = 0.f; cacc = 0.f; cur = b;
        }
        acc += h[(long long)n * 256 + c];
        cacc += 1.f;
    }
    gatomic_add_f32(sums + (long long)cur * 256 + c, acc);
    if (c == 0) gatomic_add_f32(cnts + cur, cacc);
}

// Tiny head MLP: one thread per graph (64 total, ~67K MACs each).
__global__ void head_kernel(const float* __restrict__ sums, const float* __restrict__ cnts,
                            const float* __restrict__ gattr,
                            const float* __restrict__ wf1, const float* __restrict__ bf1,
                            const float* __restrict__ wf2, const float* __restrict__ bf2,
                            float* __restrict__ out)
{
    int g = threadIdx.x;
    if (g >= NUM_GRAPHS) return;
    float inv = 1.0f / fmaxf(cnts[g], 1.0f);
    float og = bf2[0];
    for (int hcol = 0; hcol < 256; ++hcol) {
        float a = bf1[hcol];
        for (int i = 0; i < 256; ++i)
            a += sums[g * 256 + i] * inv * wf1[i * 256 + hcol];
        for (int j = 0; j < GA; ++j)
            a += gattr[g * GA + j] * wf1[(256 + j) * 256 + hcol];
        a = fmaxf(a, 0.f);
        og += a * wf2[hcol];
    }
    out[g] = og;
}

extern "C" void kernel_launch(void* const* d_in, const int* in_sizes, int n_in,
                              void* d_out, int out_size, void* d_ws, size_t ws_size,
                              hipStream_t stream)
{
    (void)in_sizes; (void)n_in; (void)out_size; (void)ws_size;
    const float* x     = (const float*)d_in[0];
    const int*   ei    = (const int*)d_in[1];
    const int*   batch = (const int*)d_in[2];
    const float* gattr = (const float*)d_in[3];
    const float* w1a = (const float*)d_in[4];  const float* b1a = (const float*)d_in[5];
    const float* w1b = (const float*)d_in[6];  const float* b1b = (const float*)d_in[7];
    const float* w2a = (const float*)d_in[8];  const float* b2a = (const float*)d_in[9];
    const float* w2b = (const float*)d_in[10]; const float* b2b = (const float*)d_in[11];
    const float* wf1 = (const float*)d_in[12]; const float* bf1 = (const float*)d_in[13];
    const float* wf2 = (const float*)d_in[14]; const float* bf2 = (const float*)d_in[15];
    const int* src = ei;
    const int* dst = ei + N_EDGES;

    // workspace layout
    float* bufP = (float*)d_ws;                                  // N x 256 f32
    float* bufQ = bufP + (long long)N_NODES * 256;               // N x 256 f32
    unsigned short* bufH = (unsigned short*)(bufQ + (long long)N_NODES * 256); // N x 256 bf16
    unsigned short* wb = bufH + (long long)N_NODES * 256;
    unsigned short* w1a_b = wb;
    unsigned short* w1b_b = wb + 128 * 256;
    unsigned short* w2a_b = wb + 128 * 256 + 256 * 256;
    unsigned short* w2b_b = wb + 128 * 256 + 2 * 256 * 256;
    float* sums = (float*)(wb + 128 * 256 + 3 * 256 * 256);      // 64 x 256
    float* cnts = sums + NUM_GRAPHS * 256;                       // 64

    // weights -> bf16 (once per launch, deterministic)
    f32_to_bf16_kernel<<<(128 * 256 / 2 + 255) / 256, 256, 0, stream>>>(
        w1a, (unsigned int*)w1a_b, 128 * 256 / 2);
    f32_to_bf16_kernel<<<(256 * 256 / 2 + 255) / 256, 256, 0, stream>>>(
        w1b, (unsigned int*)w1b_b, 256 * 256 / 2);
    f32_to_bf16_kernel<<<(256 * 256 / 2 + 255) / 256, 256, 0, stream>>>(
        w2a, (unsigned int*)w2a_b, 256 * 256 / 2);
    f32_to_bf16_kernel<<<(256 * 256 / 2 + 255) / 256, 256, 0, stream>>>(
        w2b, (unsigned int*)w2b_b, 256 * 256 / 2);

    dim3 ggrid((N_NODES + 127) / 128, 2);

    // conv1: t = x + scatter(x);  h = relu( relu(t@w1a+b1a) @ w1b + b1b )
    copy_f4<<<(N_NODES * 128 / 4) / 256, 256, 0, stream>>>((const float4*)x, (float4*)bufP,
                                                           (long long)N_NODES * 128 / 4);
    scatter_add_kernel<<<(N_EDGES * 32) / 256, 256, 0, stream>>>(x, src, dst, bufP, 5, 31, 128);
    gemm_wmma<false, true><<<ggrid, 256, 0, stream>>>(bufP, 128, 128, w1a_b, b1a,
                                                      bufH, N_NODES, 1);
    gemm_wmma<true, false><<<ggrid, 256, 0, stream>>>(bufH, 256, 256, w1b_b, b1b,
                                                      bufP, N_NODES, 1);

    // conv2: t = h + scatter(h);  out = relu(t@w2a+b2a) @ w2b + b2b
    copy_f4<<<(N_NODES * 256 / 4) / 256, 256, 0, stream>>>((const float4*)bufP, (float4*)bufQ,
                                                           (long long)N_NODES * 256 / 4);
    scatter_add_kernel<<<(N_EDGES * 64) / 256, 256, 0, stream>>>(bufP, src, dst, bufQ, 6, 63, 256);
    gemm_wmma<false, true><<<ggrid, 256, 0, stream>>>(bufQ, 256, 256, w2a_b, b2a,
                                                      bufH, N_NODES, 1);
    gemm_wmma<true, false><<<ggrid, 256, 0, stream>>>(bufH, 256, 256, w2b_b, b2b,
                                                      bufQ, N_NODES, 0);

    // mean-pool + head
    zero_f32<<<(NUM_GRAPHS * 256 + NUM_GRAPHS + 255) / 256, 256, 0, stream>>>(
        sums, NUM_GRAPHS * 256 + NUM_GRAPHS);
    pool_kernel<<<(N_NODES + 63) / 64, 256, 0, stream>>>(bufQ, batch, sums, cnts, N_NODES);
    head_kernel<<<1, 64, 0, stream>>>(sums, cnts, gattr, wf1, bf1, wf2, bf2, (float*)d_out);
}
